// sta_77275051589658
// MI455X (gfx1250) — matrix operand order repaired and checked by reading, not compiled, not run on hardware
//
#include <hip/hip_runtime.h>
#include <hip/hip_bf16.h>

// ---------------------------------------------------------------------------
// STA (temporal/channel/spatial squeeze-excite) for MI455X (gfx1250).
//
// Roofline: x is 411 MB fp32; minimal traffic = 3 reads + 1 write = 1.64 GB
// -> ~70us at 23.3 TB/s. Streaming passes are matvecs (N=1) -> VALU with
// float4 (b128) non-temporal loads/stores (x > L2's 192MB, zero reuse).
// The tiny per-branch MLPs (hidden 4096 / 2 / 256, batch N=16) are real
// 16-column GEMMs on the serial critical path -> v_wmma_f32_16x16x32_f16.
// ---------------------------------------------------------------------------

#define B_  16
#define C_  512
#define L_  16
#define HW_ 784
#define HW4_ 196          // HW/4
#define R_  256
#define RL_ 4096          // R*L (temporal hidden)
#define BC_ (B_*C_)       // 8192
#define LHW_ (L_*HW_)     // 12544
#define LHW4_ 3136        // LHW/4

typedef __attribute__((ext_vector_type(16))) _Float16 v16h;
typedef __attribute__((ext_vector_type(8)))  float    v8f;
typedef __attribute__((ext_vector_type(4)))  float    v4f;

// workspace layout (float offsets)
#define WS_TPART   0                      // B*C*L = 131072
#define WS_CPART   131072                 // B*C*L = 131072
#define WS_TSCALE  262144                 // B*L   = 256
#define WS_CVSCALE 262400                 // B*C   = 8192
#define WS_SRAW    270592                 // B*HW  = 12544
#define WS_H       283136                 // R*B   = 4096
#define WS_S2      287232                 // HW*B  = 12544
#define WS_SUMS    299776                 // 1

__device__ __forceinline__ v8f wmma_f16(v16h a, v16h b, v8f c) {
  // D = A(16x32 f16) x B(32x16 f16) + C(16x16 f32)
  return __builtin_amdgcn_wmma_f32_16x16x32_f16(
      /*neg_a=*/false, a, /*neg_b=*/false, b,
      /*c_mod=*/(short)0, c, /*reuse_a=*/false, /*reuse_b=*/false);
}

// Load a 16x32 f16 A-tile from a row-major f32 matrix W (leading dim ldw),
// rows [m0,m0+16), cols [k0,k0+kValid) (zero-padded to 32).
// A VGPR layout: lane<16 holds M=lane, K=0..15;
// lane>=16 holds M=lane-16, K=16..31; 2 f16 per VGPR.
__device__ __forceinline__ v16h load_A_f32(const float* W, int ldw,
                                           int m0, int k0, int kValid) {
  int lane = threadIdx.x & 31;
  int m = m0 + (lane & 15);
  int hiK = (lane >> 4) << 4;
  v16h a;
#pragma unroll
  for (int h = 0; h < 16; ++h) {
    int k = h + hiK;
    float v = (k < kValid) ? W[m * ldw + k0 + k] : 0.0f;
    a[h] = (_Float16)v;
  }
  return a;
}

__device__ __forceinline__ float sigmoidf_(float v) {
  return 1.0f / (1.0f + __expf(-v));
}

// ---------------------------------------------------------------------------
// Pass 1: one block per (b,c); wave w handles rows l=2w,2w+1. float4 streams.
//   Tpart[b,c,l] = sum_p x[b,c,l,p] * At3[c*HW+p]
//   Cpart[b,c,l] = sum_p x[b,c,l,p] * Ac3[l*HW+p]
// ---------------------------------------------------------------------------
__global__ void k_pass1(const float* __restrict__ x,
                        const float* __restrict__ At3,
                        const float* __restrict__ Ac3,
                        float* __restrict__ Tpart,
                        float* __restrict__ Cpart) {
  int bc   = blockIdx.x;            // b*C + c
  int c    = bc & (C_ - 1);
  int wave = threadIdx.x >> 5;
  int lane = threadIdx.x & 31;
  const v4f* xb = (const v4f*)(x + (size_t)bc * LHW_);
  const v4f* at = (const v4f*)(At3 + c * HW_);
#pragma unroll
  for (int li = 0; li < 2; ++li) {
    int l = wave * 2 + li;
    const v4f* xr = xb + l * HW4_;
    const v4f* ac = (const v4f*)(Ac3 + l * HW_);
    float st = 0.0f, sc = 0.0f;
    for (int q = lane; q < HW4_; q += 32) {       // 196 float4 per row
      v4f v = __builtin_nontemporal_load(xr + q); // x streamed, no reuse
      v4f a = at[q];
      v4f g = ac[q];
#pragma unroll
      for (int j = 0; j < 4; ++j) {
        st = fmaf(v[j], a[j], st);
        sc = fmaf(v[j], g[j], sc);
      }
    }
#pragma unroll
    for (int off = 16; off > 0; off >>= 1) {
      st += __shfl_down(st, off, 32);
      sc += __shfl_down(sc, off, 32);
    }
    if (lane == 0) {
      Tpart[bc * L_ + l] = st;
      Cpart[bc * L_ + l] = sc;
    }
  }
}

// ---------------------------------------------------------------------------
// Temporal MLP (single block, 8 waves, WMMA):
//  t_raw[l,b] = sum_c Tpart ; U = relu(Et3(4096x16) @ Traw(16x16)) ;
//  t2 = sigmoid(St3(16x4096) @ U) ; t_scale = t2 / mean(t2)
// ---------------------------------------------------------------------------
__global__ void k_tmlp(const float* __restrict__ Tpart,
                       const float* __restrict__ Et3,
                       const float* __restrict__ St3,
                       float* __restrict__ t_scale) {
  __shared__ float     traw[L_][B_];          // [l][b]
  __shared__ _Float16  reluU[8][16][16];      // per-wave [hidLocal][b]
  __shared__ float     cpart[8][256];
  __shared__ float     t2[256];
  __shared__ float     meanv;

  int tid = threadIdx.x, wave = tid >> 5, lane = tid & 31;

  { // t_raw reduction over c
    int b = tid >> 4, l = tid & 15;
    float s = 0.0f;
    for (int c = 0; c < C_; ++c) s += Tpart[(b * C_ + c) * L_ + l];
    traw[l][b] = s;
  }
  __syncthreads();

  // B operand for GEMM1: B[k=l][n=b], K=16 valid of 32.
  v16h bT;
  {
    int n = lane & 15, kb = (lane >> 4) * 16;
#pragma unroll
    for (int h = 0; h < 16; ++h) {
      int k = kb + h;
      bT[h] = (k < 16) ? (_Float16)traw[k][n] : (_Float16)0.0f;
    }
  }

  v8f c2 = {};                                  // wave partial of St3 @ relu(U)
  for (int t = 0; t < 32; ++t) {                // 8 waves x 32 tiles = 4096 rows
    int m0 = wave * 512 + t * 16;
    v16h a = load_A_f32(Et3, L_, m0, 0, 16);
    v8f u = {};
    u = wmma_f16(a, bT, u);
#pragma unroll
    for (int i = 0; i < 8; ++i) {
      float v = u[i]; v = v > 0.0f ? v : 0.0f;
      reluU[wave][i + 8 * (lane >> 4)][lane & 15] = (_Float16)v;
    }
    __syncthreads();                            // uniform; orders LDS st->ld
    v16h a2 = load_A_f32(St3, RL_, 0, m0, 16);  // St3 rows 0..15, cols m0..
    v16h b2;
    {
      int n = lane & 15, kb = (lane >> 4) * 16;
#pragma unroll
      for (int h = 0; h < 16; ++h) {
        int k = kb + h;
        b2[h] = (k < 16) ? reluU[wave][k][n] : (_Float16)0.0f;
      }
    }
    c2 = wmma_f16(a2, b2, c2);
    __syncthreads();
  }

#pragma unroll
  for (int i = 0; i < 8; ++i)
    cpart[wave][(i + 8 * (lane >> 4)) * 16 + (lane & 15)] = c2[i];
  __syncthreads();
  {
    float s = 0.0f;
    for (int w = 0; w < 8; ++w) s += cpart[w][tid];
    t2[tid] = sigmoidf_(s);                      // index = l*16 + b
  }
  __syncthreads();
  if (tid == 0) {
    float s = 0.0f;
    for (int i = 0; i < 256; ++i) s += t2[i];
    meanv = s / 256.0f;
  }
  __syncthreads();
  {
    int l = tid >> 4, b = tid & 15;
    t_scale[b * L_ + l] = t2[tid] / meanv;
  }
}

// ---------------------------------------------------------------------------
// Channel branch (single block, VALU — hidden dim is 2):
//  cv_raw[b,c] = sum_l t_scale[b,l]*Cpart[b,c,l]; hid = relu(Ec3@cv);
//  cv2 = sigmoid(Sc3@hid); cv_scale = cv2/mean(cv2). Also zeroes s_raw.
// ---------------------------------------------------------------------------
__global__ void k_cv(const float* __restrict__ Cpart,
                     const float* __restrict__ t_scale,
                     const float* __restrict__ Ec3,
                     const float* __restrict__ Sc3,
                     float* __restrict__ cv_scale,
                     float* __restrict__ s_raw) {
  __shared__ float cvr[BC_];        // 32 KB
  __shared__ float hid[B_][2];
  __shared__ float partial[256];
  int tid = threadIdx.x;

  for (int i = tid; i < BC_; i += 256) {
    int b = i >> 9;
    float s = 0.0f;
#pragma unroll
    for (int l = 0; l < L_; ++l)
      s = fmaf(t_scale[b * L_ + l], Cpart[i * L_ + l], s);
    cvr[i] = s;
  }
  for (int i = tid; i < B_ * HW_; i += 256) s_raw[i] = 0.0f;  // re-zeroed every call
  __syncthreads();

  if (tid < 32) {
    int b = tid >> 1, j = tid & 1;
    float s = 0.0f;
    for (int c = 0; c < C_; ++c) s = fmaf(Ec3[j * C_ + c], cvr[b * C_ + c], s);
    hid[b][j] = s > 0.0f ? s : 0.0f;
  }
  __syncthreads();

  float local = 0.0f;
  for (int i = tid; i < BC_; i += 256) {
    int b = i >> 9, c = i & (C_ - 1);
    float v = Sc3[c * 2 + 0] * hid[b][0] + Sc3[c * 2 + 1] * hid[b][1];
    v = sigmoidf_(v);
    cvr[i] = v;
    local += v;
  }
  partial[tid] = local;
  __syncthreads();
  if (tid == 0) {
    float s = 0.0f;
    for (int i = 0; i < 256; ++i) s += partial[i];
    partial[0] = s / (float)BC_;
  }
  __syncthreads();
  float inv = 1.0f / partial[0];
  for (int i = tid; i < BC_; i += 256) cv_scale[i] = cvr[i] * inv;
}

// ---------------------------------------------------------------------------
// Pass 2: s_raw[b,p] += sum_{c,l} x[b,c,l,p]*(t_scale*cv_scale*As3[c*L+l])
// Grid: B x (C/8) blocks; one wave per channel; float4 x streams; LDS
// (ds_add_f32) block reduction; one global atomic per pixel per block.
// ---------------------------------------------------------------------------
__global__ void k_pass2(const float* __restrict__ x,
                        const float* __restrict__ As3,
                        const float* __restrict__ t_scale,
                        const float* __restrict__ cv_scale,
                        float* __restrict__ s_raw) {
  int b  = blockIdx.x >> 6;
  int c0 = (blockIdx.x & 63) * 8;
  __shared__ float w[8][L_];
  __shared__ float sblk[HW_];
  int tid = threadIdx.x, wave = tid >> 5, lane = tid & 31;
  if (tid < 128) {
    int ci = tid >> 4, l = tid & 15, c = c0 + ci;
    w[ci][l] = t_scale[b * L_ + l] * cv_scale[b * C_ + c] * As3[c * L_ + l];
  }
  for (int i = tid; i < HW_; i += 256) sblk[i] = 0.0f;
  __syncthreads();

  int c = c0 + wave;                                  // one wave per channel
  const v4f* xc = (const v4f*)(x + ((size_t)(b * C_ + c) * L_) * HW_);
  for (int q = lane; q < HW4_; q += 32) {
    v4f acc = {0.0f, 0.0f, 0.0f, 0.0f};
#pragma unroll
    for (int l = 0; l < L_; ++l) {
      v4f v = __builtin_nontemporal_load(xc + l * HW4_ + q);
      acc += v * w[wave][l];
    }
#pragma unroll
    for (int j = 0; j < 4; ++j) atomicAdd(&sblk[q * 4 + j], acc[j]);
  }
  __syncthreads();
  for (int i = tid; i < HW_; i += 256) atomicAdd(&s_raw[b * HW_ + i], sblk[i]);
}

// ---------------------------------------------------------------------------
// Spatial MLP stage 1 (WMMA): H(256x16) = relu(Es3(256x784) @ Sraw(784x16))
// ---------------------------------------------------------------------------
__global__ void k_smlp1(const float* __restrict__ Es3,
                        const float* __restrict__ s_raw,
                        float* __restrict__ H,
                        float* __restrict__ sum_s) {
  __shared__ float part[8][256];
  int tid = threadIdx.x, wave = tid >> 5, lane = tid & 31;
  int m0 = blockIdx.x * 16;
  v8f acc = {};
  for (int chunk = wave; chunk < 25; chunk += 8) {
    int k0 = chunk * 32;
    int kValid = HW_ - k0; if (kValid > 32) kValid = 32;   // last chunk = 16
    v16h a = load_A_f32(Es3, HW_, m0, k0, kValid);
    v16h b;
    {
      int n = lane & 15, kb = (lane >> 4) * 16;
#pragma unroll
      for (int h = 0; h < 16; ++h) {
        int k = kb + h;
        b[h] = (k < kValid) ? (_Float16)s_raw[n * HW_ + k0 + k] : (_Float16)0.0f;
      }
    }
    acc = wmma_f16(a, b, acc);
  }
#pragma unroll
  for (int i = 0; i < 8; ++i)
    part[wave][(i + 8 * (lane >> 4)) * 16 + (lane & 15)] = acc[i];
  __syncthreads();
  {
    float s = 0.0f;
    for (int w = 0; w < 8; ++w) s += part[w][tid];
    s = s > 0.0f ? s : 0.0f;
    int r = m0 + (tid >> 4), b = tid & 15;
    H[r * B_ + b] = s;
  }
  if (blockIdx.x == 0 && tid == 0) *sum_s = 0.0f;  // zeroed before k_smlp2 runs
}

// ---------------------------------------------------------------------------
// Spatial MLP stage 2 (WMMA): s2(784x16) = sigmoid(Ss3(784x256) @ H(256x16))
// ---------------------------------------------------------------------------
__global__ void k_smlp2(const float* __restrict__ Ss3,
                        const float* __restrict__ H,
                        float* __restrict__ s2,
                        float* __restrict__ sum_s) {
  __shared__ float part[8][256];
  int tid = threadIdx.x, wave = tid >> 5, lane = tid & 31;
  int m0 = blockIdx.x * 16;                     // p rows
  int k0 = wave * 32;
  v16h a = load_A_f32(Ss3, R_, m0, k0, 32);
  v16h b;
  {
    int n = lane & 15, kb = (lane >> 4) * 16;
#pragma unroll
    for (int h = 0; h < 16; ++h)
      b[h] = (_Float16)H[(k0 + kb + h) * B_ + n];
  }
  v8f acc = {};
  acc = wmma_f16(a, b, acc);
#pragma unroll
  for (int i = 0; i < 8; ++i)
    part[wave][(i + 8 * (lane >> 4)) * 16 + (lane & 15)] = acc[i];
  __syncthreads();
  {
    float s = 0.0f;
    for (int w = 0; w < 8; ++w) s += part[w][tid];
    float sg = sigmoidf_(s);
    int p = m0 + (tid >> 4), bb = tid & 15;
    s2[p * B_ + bb] = sg;
    part[0][tid] = sg;                          // reuse for block mean
  }
  __syncthreads();
  if (tid == 0) {
    float s = 0.0f;
    for (int i = 0; i < 256; ++i) s += part[0][i];
    atomicAdd(sum_s, s);
  }
}

// ---------------------------------------------------------------------------
// Pass 3: out = x * t_scale[b,l] * cv_scale[b,c] * (s2[p,b] * 12544/sum_s)
// One block per (b,c); flat float4 loop (rows are multiples of 4, so a
// float4 never crosses an l-row); scales staged in LDS; NT b128 load/store.
// ---------------------------------------------------------------------------
__global__ void k_pass3(const float* __restrict__ x,
                        const float* __restrict__ t_scale,
                        const float* __restrict__ cv_scale,
                        const float* __restrict__ s2,
                        const float* __restrict__ sum_s,
                        float* __restrict__ out) {
  __shared__ v4f  ss4[HW4_];        // spatial scale, vectorized
  __shared__ float tf[L_];          // t_scale[b,:] * cv_scale[b,c]
  int bc = blockIdx.x;
  int b = bc >> 9, c = bc & (C_ - 1);
  int tid = threadIdx.x;
  float inv_mean_s = (float)(B_ * HW_) / (*sum_s);
  for (int q = tid; q < HW4_; q += 256) {
    v4f v;
#pragma unroll
    for (int j = 0; j < 4; ++j) v[j] = s2[(q * 4 + j) * B_ + b] * inv_mean_s;
    ss4[q] = v;
  }
  if (tid < L_) tf[tid] = t_scale[b * L_ + tid] * cv_scale[b * C_ + c];
  __syncthreads();

  const v4f* xi = (const v4f*)(x + (size_t)bc * LHW_);
  v4f*       oo = (v4f*)(out + (size_t)bc * LHW_);
  for (int i = tid; i < LHW4_; i += 256) {
    int l  = i / HW4_;
    int q  = i - l * HW4_;
    v4f v  = __builtin_nontemporal_load(xi + i);
    v4f r  = v * tf[l] * ss4[q];
    __builtin_nontemporal_store(r, oo + i);
  }
}

// ---------------------------------------------------------------------------
extern "C" void kernel_launch(void* const* d_in, const int* in_sizes, int n_in,
                              void* d_out, int out_size, void* d_ws, size_t ws_size,
                              hipStream_t stream) {
  (void)in_sizes; (void)n_in; (void)out_size; (void)ws_size;
  const float* x   = (const float*)d_in[0];
  const float* At3 = (const float*)d_in[1];
  const float* Et3 = (const float*)d_in[2];
  const float* St3 = (const float*)d_in[3];
  const float* Ac3 = (const float*)d_in[4];
  const float* Ec3 = (const float*)d_in[5];
  const float* Sc3 = (const float*)d_in[6];
  const float* As3 = (const float*)d_in[7];
  const float* Es3 = (const float*)d_in[8];
  const float* Ss3 = (const float*)d_in[9];
  float* out = (float*)d_out;
  float* ws  = (float*)d_ws;

  float* Tpart   = ws + WS_TPART;
  float* Cpart   = ws + WS_CPART;
  float* t_scale = ws + WS_TSCALE;
  float* cv_sc   = ws + WS_CVSCALE;
  float* s_raw   = ws + WS_SRAW;
  float* H       = ws + WS_H;
  float* s2      = ws + WS_S2;
  float* sum_s   = ws + WS_SUMS;

  k_pass1<<<BC_, 256, 0, stream>>>(x, At3, Ac3, Tpart, Cpart);
  k_tmlp <<<1,   256, 0, stream>>>(Tpart, Et3, St3, t_scale);
  k_cv   <<<1,   256, 0, stream>>>(Cpart, t_scale, Ec3, Sc3, cv_sc, s_raw);
  k_pass2<<<B_ * 64, 256, 0, stream>>>(x, As3, t_scale, cv_sc, s_raw);
  k_smlp1<<<16,  256, 0, stream>>>(Es3, s_raw, H, sum_s);
  k_smlp2<<<49,  256, 0, stream>>>(Ss3, H, s2, sum_s);
  k_pass3<<<BC_, 256, 0, stream>>>(x, t_scale, cv_sc, s2, sum_s, out);
}
